// GNNPlus_472446402723
// MI455X (gfx1250) — matrix operand order repaired and checked by reading, not compiled, not run on hardware
//
#include <hip/hip_runtime.h>
#include <hip/hip_bf16.h>
#include <stdint.h>

typedef __bf16 bf16;
typedef __bf16 v16bf __attribute__((ext_vector_type(16)));
typedef float  v8f   __attribute__((ext_vector_type(8)));

#define N_NODES 200000
#define N_EDGES 800000
#define N_GRAPH 8000
#define HDIM    256

// ---------------------------------------------------------------------------
// Pack a row-major f32 weight matrix W[K][Ncols] into the per-lane WMMA
// B-fragment layout (16x16x32 bf16): fragment index f = (ct*(K/32)+ks)*32+lane,
// each fragment = 16 consecutive bf16 (32B) so a wave loads it with one
// contiguous v16bf (2x global_load_b128) per lane.
// B layout (32x16, bf16): lane holds col n = ct*16 + (lane&15);
// VGPR v holds K = ks*32 + 2v + (e&1) + (v>=4 ? 8 : 0) + (lane>=16 ? 8 : 0).
// ---------------------------------------------------------------------------
__global__ void pack_w_kernel(const float* __restrict__ W, bf16* __restrict__ P,
                              int K, int Ncols) {
    int tid = blockIdx.x * blockDim.x + threadIdx.x;
    int total = K * Ncols;
    if (tid >= total) return;
    int e    = tid & 15;
    int l    = (tid >> 4) & 31;
    int rest = tid >> 9;
    int KS   = K >> 5;
    int ks   = rest % KS;
    int ct   = rest / KS;
    int n    = ct * 16 + (l & 15);
    int v    = e >> 1;
    int k    = ks * 32 + 2 * v + (e & 1) + ((v >= 4) ? 8 : 0) + ((l >= 16) ? 8 : 0);
    P[tid] = (bf16)W[(size_t)k * Ncols + n];
}

// ---------------------------------------------------------------------------
// Row-wise 2-layer MLP:  out = act2( relu( X @ W1 + b1 ) @ W2 + b2 )
// X: [rows, IN_DIM] f32.  W1p/W2p: packed bf16 fragments.
// 128 threads = 4 waves per block; 32 rows per wave (two 16-row A tiles that
// share every B fragment -> 2 WMMA per 32B weight load, half the L2->L0
// weight traffic); 128 rows per block. Hidden staged in LDS as bf16 (64 KB).
// ---------------------------------------------------------------------------
template<int IN_DIM, bool OUT_BF16, bool RELU_OUT>
__launch_bounds__(128)
__global__ void mlp2_wmma_kernel(const float* __restrict__ X, int rows,
                                 const bf16* __restrict__ W1p,
                                 const float* __restrict__ b1,
                                 const bf16* __restrict__ W2p,
                                 const float* __restrict__ b2,
                                 void* __restrict__ outv) {
    __shared__ bf16 hid[4][2][16][HDIM];   // 64 KB

    const int wave = threadIdx.x >> 5;
    const int lane = threadIdx.x & 31;
    const int m    = lane & 15;               // A row-within-tile / C col-within-tile
    const int half = (lane >= 16) ? 8 : 0;
    const int rowBase = blockIdx.x * 128 + wave * 32;

    int arow0 = rowBase + m;
    int arow1 = rowBase + 16 + m;
    if (arow0 > rows - 1) arow0 = rows - 1;   // clamp for ragged last block
    if (arow1 > rows - 1) arow1 = rows - 1;
    const float* xr0 = X + (size_t)arow0 * IN_DIM;
    const float* xr1 = X + (size_t)arow1 * IN_DIM;

    // ---- Phase 1: hidden = relu(X @ W1 + b1), kept as bf16 in LDS ----
    constexpr int KS1 = IN_DIM / 32;
    v16bf a1[2][KS1];
#pragma unroll
    for (int ks = 0; ks < KS1; ++ks) {
#pragma unroll
        for (int vv = 0; vv < 8; ++vv) {
            int k0 = ks * 32 + 2 * vv + ((vv >= 4) ? 8 : 0) + half;
            a1[0][ks][2 * vv]     = (bf16)xr0[k0];
            a1[0][ks][2 * vv + 1] = (bf16)xr0[k0 + 1];
            a1[1][ks][2 * vv]     = (bf16)xr1[k0];
            a1[1][ks][2 * vv + 1] = (bf16)xr1[k0 + 1];
        }
    }

    const v16bf* W1f = (const v16bf*)W1p;
#pragma unroll 1
    for (int ct = 0; ct < 16; ++ct) {
        v8f acc0 = {}, acc1 = {};
#pragma unroll
        for (int ks = 0; ks < KS1; ++ks) {
            v16bf b = W1f[(ct * KS1 + ks) * 32 + lane];
            acc0 = __builtin_amdgcn_wmma_f32_16x16x32_bf16(
                       false, a1[0][ks], false, b, (short)0, acc0, false, false);
            acc1 = __builtin_amdgcn_wmma_f32_16x16x32_bf16(
                       false, a1[1][ks], false, b, (short)0, acc1, false, false);
        }
        float bias = b1[ct * 16 + m];
#pragma unroll
        for (int r = 0; r < 8; ++r) {
            float v0 = acc0[r] + bias;
            float v1 = acc1[r] + bias;
            v0 = v0 > 0.f ? v0 : 0.f;
            v1 = v1 > 0.f ? v1 : 0.f;
            hid[wave][0][r + half][ct * 16 + m] = (bf16)v0;
            hid[wave][1][r + half][ct * 16 + m] = (bf16)v1;
        }
    }
    __syncthreads();

    // ---- Phase 2: out = hidden @ W2 + b2 (+ optional relu) ----
    v16bf a2[2][8];
#pragma unroll
    for (int ks = 0; ks < 8; ++ks) {
#pragma unroll
        for (int vv = 0; vv < 8; ++vv) {
            int k0 = ks * 32 + 2 * vv + ((vv >= 4) ? 8 : 0) + half;
            a2[0][ks][2 * vv]     = hid[wave][0][m][k0];
            a2[0][ks][2 * vv + 1] = hid[wave][0][m][k0 + 1];
            a2[1][ks][2 * vv]     = hid[wave][1][m][k0];
            a2[1][ks][2 * vv + 1] = hid[wave][1][m][k0 + 1];
        }
    }

    const v16bf* W2f = (const v16bf*)W2p;
#pragma unroll 1
    for (int ct = 0; ct < 16; ++ct) {
        v8f acc0 = {}, acc1 = {};
#pragma unroll
        for (int ks = 0; ks < 8; ++ks) {
            v16bf b = W2f[(ct * 8 + ks) * 32 + lane];
            acc0 = __builtin_amdgcn_wmma_f32_16x16x32_bf16(
                       false, a2[0][ks], false, b, (short)0, acc0, false, false);
            acc1 = __builtin_amdgcn_wmma_f32_16x16x32_bf16(
                       false, a2[1][ks], false, b, (short)0, acc1, false, false);
        }
        float bias = b2[ct * 16 + m];
#pragma unroll
        for (int r = 0; r < 8; ++r) {
            int rr0 = rowBase + r + half;
            int rr1 = rowBase + 16 + r + half;
            float v0 = acc0[r] + bias;
            float v1 = acc1[r] + bias;
            if (RELU_OUT) { v0 = v0 > 0.f ? v0 : 0.f; v1 = v1 > 0.f ? v1 : 0.f; }
            if (rr0 < rows) {
                if (OUT_BF16) ((bf16*)outv)[(size_t)rr0 * HDIM + ct * 16 + m] = (bf16)v0;
                else          ((float*)outv)[(size_t)rr0 * HDIM + ct * 16 + m] = v0;
            }
            if (rr1 < rows) {
                if (OUT_BF16) ((bf16*)outv)[(size_t)rr1 * HDIM + ct * 16 + m] = (bf16)v1;
                else          ((float*)outv)[(size_t)rr1 * HDIM + ct * 16 + m] = v1;
            }
        }
    }
}

// ---------------------------------------------------------------------------
// agg[dst] += relu(feat[src] + ea)   (64 threads x float4 per edge row)
// ---------------------------------------------------------------------------
__global__ void msg_scatter_kernel(const float* __restrict__ feat,
                                   const bf16* __restrict__ ea,
                                   const int* __restrict__ src,
                                   const int* __restrict__ dst,
                                   float* __restrict__ agg, int nE) {
    long t = (long)blockIdx.x * blockDim.x + threadIdx.x;
    int e = (int)(t >> 6);
    if (e >= nE) return;
    int c = ((int)t & 63) * 4;
    int s = src[e], d = dst[e];
    const float4 xv = *(const float4*)(feat + (size_t)s * HDIM + c);
    const bf16* ep = ea + (size_t)e * HDIM + c;
    float m0 = xv.x + (float)ep[0];
    float m1 = xv.y + (float)ep[1];
    float m2 = xv.z + (float)ep[2];
    float m3 = xv.w + (float)ep[3];
    m0 = m0 > 0.f ? m0 : 0.f;
    m1 = m1 > 0.f ? m1 : 0.f;
    m2 = m2 > 0.f ? m2 : 0.f;
    m3 = m3 > 0.f ? m3 : 0.f;
    float* ap = agg + (size_t)d * HDIM + c;
    __hip_atomic_fetch_add(ap + 0, m0, __ATOMIC_RELAXED, __HIP_MEMORY_SCOPE_AGENT);
    __hip_atomic_fetch_add(ap + 1, m1, __ATOMIC_RELAXED, __HIP_MEMORY_SCOPE_AGENT);
    __hip_atomic_fetch_add(ap + 2, m2, __ATOMIC_RELAXED, __HIP_MEMORY_SCOPE_AGENT);
    __hip_atomic_fetch_add(ap + 3, m3, __ATOMIC_RELAXED, __HIP_MEMORY_SCOPE_AGENT);
}

__global__ void copy_f4_kernel(const float4* __restrict__ s, float4* __restrict__ d, long n) {
    long i = (long)blockIdx.x * blockDim.x + threadIdx.x;
    if (i < n) d[i] = s[i];
}

__global__ void zero_kernel(float* __restrict__ p, long n) {
    long i = (long)blockIdx.x * blockDim.x + threadIdx.x;
    if (i < n) p[i] = 0.f;
}

// sums[batch[n]] += h[n];  cnts[batch[n]] += 1
__global__ void pool_accum_kernel(const float* __restrict__ h,
                                  const int* __restrict__ batch,
                                  float* __restrict__ sums,
                                  float* __restrict__ cnts, int nN) {
    long t = (long)blockIdx.x * blockDim.x + threadIdx.x;
    int n = (int)(t >> 6);
    if (n >= nN) return;
    int c = ((int)t & 63) * 4;
    int g = batch[n];
    const float4 v = *(const float4*)(h + (size_t)n * HDIM + c);
    float* sp = sums + (size_t)g * HDIM + c;
    __hip_atomic_fetch_add(sp + 0, v.x, __ATOMIC_RELAXED, __HIP_MEMORY_SCOPE_AGENT);
    __hip_atomic_fetch_add(sp + 1, v.y, __ATOMIC_RELAXED, __HIP_MEMORY_SCOPE_AGENT);
    __hip_atomic_fetch_add(sp + 2, v.z, __ATOMIC_RELAXED, __HIP_MEMORY_SCOPE_AGENT);
    __hip_atomic_fetch_add(sp + 3, v.w, __ATOMIC_RELAXED, __HIP_MEMORY_SCOPE_AGENT);
    if (((int)t & 63) == 0)
        __hip_atomic_fetch_add(cnts + g, 1.0f, __ATOMIC_RELAXED, __HIP_MEMORY_SCOPE_AGENT);
}

// Per-graph head MLP: g = relu(relu(pooled@l1+b)@l2+b); 3 scalar heads.
__launch_bounds__(128)
__global__ void heads_kernel(const float* __restrict__ sums, const float* __restrict__ cnts,
                             const float* __restrict__ l1w, const float* __restrict__ l1b,
                             const float* __restrict__ l2w, const float* __restrict__ l2b,
                             const float* __restrict__ hSw, const float* __restrict__ hSb,
                             const float* __restrict__ hPw, const float* __restrict__ hPb,
                             const float* __restrict__ hNw, const float* __restrict__ hNb,
                             float* __restrict__ out, int nG) {
    int g = blockIdx.x;
    if (g >= nG) return;
    __shared__ float pooled[256];
    __shared__ float g1[128];
    __shared__ float g2[64];
    int t = threadIdx.x;  // 128 threads
    float inv = 1.0f / fmaxf(cnts[g], 1.0f);
    pooled[t]       = sums[(size_t)g * HDIM + t] * inv;
    pooled[t + 128] = sums[(size_t)g * HDIM + t + 128] * inv;
    __syncthreads();
    float a = l1b[t];
    for (int k = 0; k < 256; ++k) a += pooled[k] * l1w[(size_t)k * 128 + t];
    g1[t] = fmaxf(a, 0.f);
    __syncthreads();
    if (t < 64) {
        float b = l2b[t];
        for (int k = 0; k < 128; ++k) b += g1[k] * l2w[(size_t)k * 64 + t];
        g2[t] = fmaxf(b, 0.f);
    }
    __syncthreads();
    if (t == 0) {
        float s = hSb[0], p = hPb[0], q = hNb[0];
        for (int k = 0; k < 64; ++k) {
            float v = g2[k];
            s += v * hSw[k];
            p += v * hPw[k];
            q += v * hNw[k];
        }
        out[g]          = s;
        out[nG + g]     = p;
        out[2 * nG + g] = q;
    }
}

// ---------------------------------------------------------------------------
extern "C" void kernel_launch(void* const* d_in, const int* in_sizes, int n_in,
                              void* d_out, int out_size, void* d_ws, size_t ws_size,
                              hipStream_t stream) {
    const float* x         = (const float*)d_in[0];
    const float* edge_attr = (const float*)d_in[1];
    const int*   eidx      = (const int*)d_in[2];
    const int*   batch     = (const int*)d_in[3];
    const float* em_w1 = (const float*)d_in[4];  const float* em_b1 = (const float*)d_in[5];
    const float* em_w2 = (const float*)d_in[6];  const float* em_b2 = (const float*)d_in[7];
    const float* c1_w1 = (const float*)d_in[8];  const float* c1_b1 = (const float*)d_in[9];
    const float* c1_w2 = (const float*)d_in[10]; const float* c1_b2 = (const float*)d_in[11];
    const float* c2_w1 = (const float*)d_in[12]; const float* c2_b1 = (const float*)d_in[13];
    const float* c2_w2 = (const float*)d_in[14]; const float* c2_b2 = (const float*)d_in[15];
    const float* l1_w  = (const float*)d_in[16]; const float* l1_b  = (const float*)d_in[17];
    const float* l2_w  = (const float*)d_in[18]; const float* l2_b  = (const float*)d_in[19];
    const float* hS_w  = (const float*)d_in[20]; const float* hS_b  = (const float*)d_in[21];
    const float* hP_w  = (const float*)d_in[22]; const float* hP_b  = (const float*)d_in[23];
    const float* hN_w  = (const float*)d_in[24]; const float* hN_b  = (const float*)d_in[25];

    const int* src = eidx;
    const int* dst = eidx + N_EDGES;

    char* ws = (char*)d_ws;
    size_t off = 0;
    auto take = [&](size_t bytes) -> char* {
        char* p = ws + off;
        off = (off + bytes + 255) & ~(size_t)255;
        return p;
    };
    bf16*  ea   = (bf16*) take((size_t)N_EDGES * HDIM * 2);   // edge MLP output (bf16)
    float* bufA = (float*)take((size_t)N_NODES * HDIM * 4);   // x+agg accumulator
    float* bufB = (float*)take((size_t)N_NODES * HDIM * 4);   // conv output
    bf16*  pem1 = (bf16*) take((size_t)64  * 256 * 2);
    bf16*  pem2 = (bf16*) take((size_t)256 * 256 * 2);
    bf16*  pc11 = (bf16*) take((size_t)256 * 256 * 2);
    bf16*  pc12 = (bf16*) take((size_t)256 * 256 * 2);
    bf16*  pc21 = (bf16*) take((size_t)256 * 256 * 2);
    bf16*  pc22 = (bf16*) take((size_t)256 * 256 * 2);
    float* sums = (float*)take((size_t)N_GRAPH * HDIM * 4);
    float* cnts = (float*)take((size_t)N_GRAPH * 4);

    // 1) pack weights to bf16 WMMA B-fragment layout
    pack_w_kernel<<<(64 * 256 + 255) / 256, 256, 0, stream>>>(em_w1, pem1, 64, 256);
    pack_w_kernel<<<(256 * 256 + 255) / 256, 256, 0, stream>>>(em_w2, pem2, 256, 256);
    pack_w_kernel<<<(256 * 256 + 255) / 256, 256, 0, stream>>>(c1_w1, pc11, 256, 256);
    pack_w_kernel<<<(256 * 256 + 255) / 256, 256, 0, stream>>>(c1_w2, pc12, 256, 256);
    pack_w_kernel<<<(256 * 256 + 255) / 256, 256, 0, stream>>>(c2_w1, pc21, 256, 256);
    pack_w_kernel<<<(256 * 256 + 255) / 256, 256, 0, stream>>>(c2_w2, pc22, 256, 256);

    // 2) edge MLP: ea = MLP2(edge_attr)   [E,64] -> [E,256] bf16
    mlp2_wmma_kernel<64, true, false><<<N_EDGES / 128, 128, 0, stream>>>(
        edge_attr, N_EDGES, pem1, em_b1, pem2, em_b2, ea);

    long nf4 = (long)N_NODES * HDIM / 4;
    int  cpb = (int)((nf4 + 255) / 256);
    int  msb = (int)(((long)N_EDGES * 64 + 255) / 256);
    int  ngb = (N_NODES + 127) / 128;

    // 3) conv1: bufA = x + sum relu(x[src]+ea);  h1 = relu(MLP2(bufA)) -> bufB
    copy_f4_kernel<<<cpb, 256, 0, stream>>>((const float4*)x, (float4*)bufA, nf4);
    msg_scatter_kernel<<<msb, 256, 0, stream>>>(x, ea, src, dst, bufA, N_EDGES);
    mlp2_wmma_kernel<256, false, true><<<ngb, 128, 0, stream>>>(
        bufA, N_NODES, pc11, c1_b1, pc12, c1_b2, bufB);

    // 4) conv2: bufA = h1 + sum relu(h1[src]+ea); h2 = relu(MLP2(bufA)) -> bufB
    copy_f4_kernel<<<cpb, 256, 0, stream>>>((const float4*)bufB, (float4*)bufA, nf4);
    msg_scatter_kernel<<<msb, 256, 0, stream>>>(bufB, ea, src, dst, bufA, N_EDGES);
    mlp2_wmma_kernel<256, false, true><<<ngb, 128, 0, stream>>>(
        bufA, N_NODES, pc21, c2_b1, pc22, c2_b2, bufB);

    // 5) global mean pool + heads
    zero_kernel<<<((N_GRAPH * HDIM) + 255) / 256, 256, 0, stream>>>(sums, (long)N_GRAPH * HDIM);
    zero_kernel<<<(N_GRAPH + 255) / 256, 256, 0, stream>>>(cnts, (long)N_GRAPH);
    pool_accum_kernel<<<(int)(((long)N_NODES * 64 + 255) / 256), 256, 0, stream>>>(
        bufB, batch, sums, cnts, N_NODES);
    heads_kernel<<<N_GRAPH, 128, 0, stream>>>(sums, cnts, l1_w, l1_b, l2_w, l2_b,
                                              hS_w, hS_b, hP_w, hP_b, hN_w, hN_b,
                                              (float*)d_out, N_GRAPH);
}
